// LSTMAutoencoder_41729902248192
// MI455X (gfx1250) — matrix operand order repaired and checked by reading, not compiled, not run on hardware
//
#include <hip/hip_runtime.h>

// ---------------------------------------------------------------------------
// LSTM autoencoder, fully fused, bf16 WMMA (v_wmma_f32_16x16x32_bf16).
// B=8192, T=30, F=16, H1=64, H2=32.  Each wave32 owns a 16-row batch tile.
// All weights staged in LDS (phase A / phase B overlayed) so the sequential
// recurrence chain only touches LDS + registers.
// ---------------------------------------------------------------------------

typedef __attribute__((ext_vector_type(16))) __bf16        v16bf;
typedef __attribute__((ext_vector_type(8)))  float         v8f;
typedef __attribute__((ext_vector_type(4)))  unsigned int  u32x4;

union FragBF { v16bf v; u32x4 q[2]; };
union F4     { u32x4 u; float f[4]; };

#define WAVES 4
#define T_SEQ 30

// LDS weight region offsets (ushort units), phase A overlay:
//   [ W1c 256x96 | W2c 128x96 ]              = 36864 ushorts
// phase B overlay (re-staged into same region):
//   [ W3c 128x64 | W4c 256x96 | Woc 16x64 ]  = 33792 ushorts
#define SW_W1 0
#define SW_W2 (256 * 96)
#define SW_W3 0
#define SW_W4 (128 * 64)
#define SW_WO (SW_W4 + 256 * 96)
#define SW_TOTAL (256 * 96 + 128 * 96)   // 36864 ushorts = 73728 B

__device__ __forceinline__ unsigned short f2bf(float x) {
  union { float f; unsigned int u; } cv; cv.f = x;
  unsigned int u = cv.u + 0x7fffu + ((cv.u >> 16) & 1u);   // RNE
  return (unsigned short)(u >> 16);
}

__device__ __forceinline__ float sigmoidf_(float x) {
  return 1.0f / (1.0f + __expf(-x));
}

// CDNA5 split-counter wait: all prior DS ops complete. Used for wave-local
// LDS transpose buffers (written in C/D layout, read in A layout by other
// lanes of the same wave; DS is in-order per wave, this orders the compiler
// and covers the VGPR-return hazard).
__device__ __forceinline__ void lds_fence() {
  asm volatile("s_wait_dscnt 0" ::: "memory");
}

__device__ __forceinline__ v8f wmma_bf16(v16bf a, v16bf b, v8f c) {
  return __builtin_amdgcn_wmma_f32_16x16x32_bf16(
      /*neg_a=*/false, a, /*neg_b=*/false, b,
      /*c_mod=*/(short)0, c, /*reuse_a=*/false, /*reuse_b=*/false);
}

// A fragment (16x32 bf16, M x K): lane (half,m) -> row m, K = half*8 + [0..7]
// in elements 0..7 and K = 16 + half*8 + [0..7] in elements 8..15.
__device__ __forceinline__ v16bf load_frag_a(const unsigned short* row, int halfsel) {
  FragBF f;
  f.q[0] = *(const u32x4*)(row + halfsel * 8);
  f.q[1] = *(const u32x4*)(row + 16 + halfsel * 8);
  return f.v;
}

// B fragment (32x16 bf16, K x N) from row-major W[n][k] in LDS:
// lane (half,n) element e -> K = half*8 + e (e<8), 16 + half*8 + (e-8) (e>=8).
__device__ __forceinline__ v16bf load_frag_b(const unsigned short* wrow, int halfsel) {
  FragBF f;
  f.q[0] = *(const u32x4*)(wrow + halfsel * 8);
  f.q[1] = *(const u32x4*)(wrow + 16 + halfsel * 8);
  return f.v;
}

// One LSTM step for a 16-row batch tile.
//   buf : [16][KC] bf16 LDS activations, row-major, cols = [x_t | h_{t-1} | pad0]
//   Wc  : [4*H][KC] bf16 packed weights in LDS, row n = [Wih[n] | Whh[n] | pad0]
//   h,c : persistent state in WMMA C/D layout, HT tiles of v8f
// Gate tile (G*H + q*16) and state tile q share lane/slot -> pure VALU update.
template <int KC, int HT>
__device__ __forceinline__ void lstm_step(
    const unsigned short* buf, const unsigned short* Wc,
    const float* bI, const float* bF, const float* bG, const float* bO,
    v8f* h, v8f* c, int halfsel, int n16)
{
  constexpr int NCH = KC / 32;
  constexpr int H = HT * 16;
  v16bf a[NCH];
#pragma unroll
  for (int ch = 0; ch < NCH; ++ch)
    a[ch] = load_frag_a(buf + n16 * KC + ch * 32, halfsel);

#pragma unroll
  for (int q = 0; q < HT; ++q) {
    v8f zi, zf, zg, zo;
#pragma unroll
    for (int r = 0; r < 8; ++r) { zi[r] = bI[q]; zf[r] = bF[q]; zg[r] = bG[q]; zo[r] = bO[q]; }
    const int col = q * 16 + n16;
#pragma unroll
    for (int ch = 0; ch < NCH; ++ch) {
      zi = wmma_bf16(a[ch], load_frag_b(Wc + (0 * H + col) * KC + ch * 32, halfsel), zi);
      zf = wmma_bf16(a[ch], load_frag_b(Wc + (1 * H + col) * KC + ch * 32, halfsel), zf);
      zg = wmma_bf16(a[ch], load_frag_b(Wc + (2 * H + col) * KC + ch * 32, halfsel), zg);
      zo = wmma_bf16(a[ch], load_frag_b(Wc + (3 * H + col) * KC + ch * 32, halfsel), zo);
    }
#pragma unroll
    for (int r = 0; r < 8; ++r) {
      float cn = sigmoidf_(zf[r]) * c[q][r] + sigmoidf_(zi[r]) * tanhf(zg[r]);
      c[q][r] = cn;
      h[q][r] = sigmoidf_(zo[r]) * tanhf(cn);
    }
  }
}

// Scatter h (C/D layout) into an LDS activation buffer at column col0 as bf16.
template <int KC, int HT>
__device__ __forceinline__ void store_h(unsigned short* buf, int col0, const v8f* h,
                                        int halfsel, int n16)
{
#pragma unroll
  for (int q = 0; q < HT; ++q)
#pragma unroll
    for (int r = 0; r < 8; ++r)
      buf[(r + 8 * halfsel) * KC + col0 + q * 16 + n16] = f2bf(h[q][r]);
}

// Block-cooperative: pack one LSTM layer's weights fp32->bf16 into LDS as
// [N][KC] with row n = [Wih[n](I) | Whh[n](H) | pad0].
template <int N, int KC, int I, int H>
__device__ __forceinline__ void stage_layer(unsigned short* dst,
                                            const float* __restrict__ Wih,
                                            const float* __restrict__ Whh,
                                            int tid, int nthreads)
{
  for (int i = tid; i < N * KC; i += nthreads) {
    const int n = i / KC, k = i % KC;
    float v = (k < I) ? Wih[n * I + k]
                      : ((k < I + H) ? Whh[n * H + (k - I)] : 0.0f);
    dst[i] = f2bf(v);
  }
}

// ---------------------------------------------------------------------------
// Fused kernel: 4 waves/block, each wave = 16 batch rows, waves independent
// except for cooperative weight staging. Phase A: layers 1+2 over t.
// Phase B: layers 3+4 + output projection over t.
// ---------------------------------------------------------------------------
__global__ __launch_bounds__(32 * WAVES) void lstm_ae_fused_kernel(
    const float* __restrict__ x,
    const float* __restrict__ Wih1, const float* __restrict__ Whh1, const float* __restrict__ b1,
    const float* __restrict__ Wih2, const float* __restrict__ Whh2, const float* __restrict__ b2,
    const float* __restrict__ Wih3, const float* __restrict__ Whh3, const float* __restrict__ b3,
    const float* __restrict__ Wih4, const float* __restrict__ Whh4, const float* __restrict__ b4,
    const float* __restrict__ Wout, const float* __restrict__ bout,
    float* __restrict__ out)
{
  __shared__ unsigned short sW[SW_TOTAL];                    // 73728 B, overlayed per phase
  __shared__ unsigned short sAct[WAVES][16 * 96 * 3 + 16 * 64];  // 45056 B

  const int tid = threadIdx.x;
  const int lane = tid & 31;
  const int wave = tid >> 5;
  const int halfsel = lane >> 4;     // K half for A/B fragments, M half for C/D
  const int n16 = lane & 15;         // N (and batch-row for A-fragment loads)
  const int b0 = (blockIdx.x * WAVES + wave) * 16;

  unsigned short* buf1 = sAct[wave];                 // [16][96]: x(16)|h1(64)|pad(16)
  unsigned short* buf2 = buf1 + 16 * 96;             // [16][96]: h1(64)|h2(32)
  unsigned short* buf3 = buf2 + 16 * 96;             // [16][64]: latent(32)|h3(32)
  unsigned short* buf4 = buf3 + 16 * 64;             // [16][96]: h3(32)|h4(64)

  // ---- stage phase-A weights (block-cooperative) + zero activation buffers
  stage_layer<256, 96, 16, 64>(sW + SW_W1, Wih1, Whh1, tid, 32 * WAVES);
  stage_layer<128, 96, 64, 32>(sW + SW_W2, Wih2, Whh2, tid, 32 * WAVES);
  {
    unsigned int* p = (unsigned int*)sAct[wave];
    constexpr int NW = (16 * 96 * 3 + 16 * 64) / 2;   // 2816 uints, 88 per lane
#pragma unroll 4
    for (int i = lane; i < NW; i += 32) p[i] = 0u;
  }
  __syncthreads();

  // phase-A biases (per-lane, one per 16-wide hidden tile)
  float bI1[4], bF1[4], bG1[4], bO1[4];
#pragma unroll
  for (int q = 0; q < 4; ++q) {
    bI1[q] = b1[0 + q * 16 + n16];   bF1[q] = b1[64 + q * 16 + n16];
    bG1[q] = b1[128 + q * 16 + n16]; bO1[q] = b1[192 + q * 16 + n16];
  }
  float bI2[2], bF2[2], bG2[2], bO2[2];
#pragma unroll
  for (int q = 0; q < 2; ++q) {
    bI2[q] = b2[0 + q * 16 + n16];  bF2[q] = b2[32 + q * 16 + n16];
    bG2[q] = b2[64 + q * 16 + n16]; bO2[q] = b2[96 + q * 16 + n16];
  }

  v8f h1[4], c1[4], h2[2], c2[2];
#pragma unroll
  for (int q = 0; q < 4; ++q)
#pragma unroll
    for (int r = 0; r < 8; ++r) { h1[q][r] = 0.0f; c1[q][r] = 0.0f; }
#pragma unroll
  for (int q = 0; q < 2; ++q)
#pragma unroll
    for (int r = 0; r < 8; ++r) { h2[q][r] = 0.0f; c2[q][r] = 0.0f; }

  // ------------------------- Phase A: encoder -------------------------
#pragma unroll 1
  for (int t = 0; t < T_SEQ; ++t) {
    // stage x_t: lane (half,row) loads 8 floats, packs to bf16, one b128 store
    {
      const float* xp = x + ((size_t)(b0 + n16) * T_SEQ + t) * 16 + halfsel * 8;
      F4 a0, a1;
      a0.u = *(const u32x4*)xp;
      a1.u = *(const u32x4*)(xp + 4);
      u32x4 pk;
      pk[0] = (unsigned int)f2bf(a0.f[0]) | ((unsigned int)f2bf(a0.f[1]) << 16);
      pk[1] = (unsigned int)f2bf(a0.f[2]) | ((unsigned int)f2bf(a0.f[3]) << 16);
      pk[2] = (unsigned int)f2bf(a1.f[0]) | ((unsigned int)f2bf(a1.f[1]) << 16);
      pk[3] = (unsigned int)f2bf(a1.f[2]) | ((unsigned int)f2bf(a1.f[3]) << 16);
      *(u32x4*)(buf1 + n16 * 96 + halfsel * 8) = pk;
    }
    lds_fence();

    lstm_step<96, 4>(buf1, sW + SW_W1, bI1, bF1, bG1, bO1, h1, c1, halfsel, n16);
    store_h<96, 4>(buf1, 16, h1, halfsel, n16);   // h1 -> own recurrence input
    store_h<96, 4>(buf2, 0, h1, halfsel, n16);    // h1 -> layer2 input (= ys1[t])
    lds_fence();

    lstm_step<96, 2>(buf2, sW + SW_W2, bI2, bF2, bG2, bO2, h2, c2, halfsel, n16);
    store_h<96, 2>(buf2, 64, h2, halfsel, n16);   // h2 -> own recurrence input
    lds_fence();
  }

  // latent = h2 -> buf3 cols 0..31 (constant layer3 input for every t)
  store_h<64, 2>(buf3, 0, h2, halfsel, n16);
  lds_fence();

  // ---- re-stage weight region with phase-B weights (all waves must be done)
  __syncthreads();
  stage_layer<128, 64, 32, 32>(sW + SW_W3, Wih3, Whh3, tid, 32 * WAVES);
  stage_layer<256, 96, 32, 64>(sW + SW_W4, Wih4, Whh4, tid, 32 * WAVES);
  for (int i = tid; i < 16 * 64; i += 32 * WAVES) sW[SW_WO + i] = f2bf(Wout[i]);
  __syncthreads();

  // phase-B biases
  float bI3[2], bF3[2], bG3[2], bO3[2];
#pragma unroll
  for (int q = 0; q < 2; ++q) {
    bI3[q] = b3[0 + q * 16 + n16];  bF3[q] = b3[32 + q * 16 + n16];
    bG3[q] = b3[64 + q * 16 + n16]; bO3[q] = b3[96 + q * 16 + n16];
  }
  float bI4[4], bF4[4], bG4[4], bO4[4];
#pragma unroll
  for (int q = 0; q < 4; ++q) {
    bI4[q] = b4[0 + q * 16 + n16];   bF4[q] = b4[64 + q * 16 + n16];
    bG4[q] = b4[128 + q * 16 + n16]; bO4[q] = b4[192 + q * 16 + n16];
  }
  const float bo = bout[n16];

  v8f h3[2], c3[2], h4[4], c4[4];
#pragma unroll
  for (int q = 0; q < 2; ++q)
#pragma unroll
    for (int r = 0; r < 8; ++r) { h3[q][r] = 0.0f; c3[q][r] = 0.0f; }
#pragma unroll
  for (int q = 0; q < 4; ++q)
#pragma unroll
    for (int r = 0; r < 8; ++r) { h4[q][r] = 0.0f; c4[q][r] = 0.0f; }

  // --------------------- Phase B: decoder + projection ---------------------
#pragma unroll 1
  for (int t = 0; t < T_SEQ; ++t) {
    lstm_step<64, 2>(buf3, sW + SW_W3, bI3, bF3, bG3, bO3, h3, c3, halfsel, n16);
    store_h<64, 2>(buf3, 32, h3, halfsel, n16);   // h3 -> own recurrence input
    store_h<96, 2>(buf4, 0, h3, halfsel, n16);    // h3 -> layer4 input
    lds_fence();

    lstm_step<96, 4>(buf4, sW + SW_W4, bI4, bF4, bG4, bO4, h4, c4, halfsel, n16);
    store_h<96, 4>(buf4, 32, h4, halfsel, n16);   // h4 -> recurrence + projection A
    lds_fence();

    // out[:, t, :] = h4 @ Wout^T + bout   (one 16x16 tile, K = 64)
    v8f acc;
#pragma unroll
    for (int r = 0; r < 8; ++r) acc[r] = bo;
#pragma unroll
    for (int ch = 0; ch < 2; ++ch) {
      v16bf a = load_frag_a(buf4 + n16 * 96 + 32 + ch * 32, halfsel);
      acc = wmma_bf16(a, load_frag_b(sW + SW_WO + n16 * 64 + ch * 32, halfsel), acc);
    }
#pragma unroll
    for (int r = 0; r < 8; ++r) {
      const int row = r + 8 * halfsel;
      out[((size_t)(b0 + row) * T_SEQ + t) * 16 + n16] = acc[r];
    }
  }
}

// ---------------------------------------------------------------------------
extern "C" void kernel_launch(void* const* d_in, const int* in_sizes, int n_in,
                              void* d_out, int out_size, void* d_ws, size_t ws_size,
                              hipStream_t stream) {
  const float* x    = (const float*)d_in[0];
  const float* Wih1 = (const float*)d_in[1];
  const float* Whh1 = (const float*)d_in[2];
  const float* b1   = (const float*)d_in[3];
  const float* Wih2 = (const float*)d_in[4];
  const float* Whh2 = (const float*)d_in[5];
  const float* b2   = (const float*)d_in[6];
  const float* Wih3 = (const float*)d_in[7];
  const float* Whh3 = (const float*)d_in[8];
  const float* b3   = (const float*)d_in[9];
  const float* Wih4 = (const float*)d_in[10];
  const float* Whh4 = (const float*)d_in[11];
  const float* b4   = (const float*)d_in[12];
  const float* Wout = (const float*)d_in[13];
  const float* bout = (const float*)d_in[14];
  (void)d_ws; (void)ws_size; (void)in_sizes; (void)n_in; (void)out_size;

  const int B = 8192;
  lstm_ae_fused_kernel<<<dim3(B / (16 * WAVES)), dim3(32 * WAVES), 0, stream>>>(
      x, Wih1, Whh1, b1, Wih2, Whh2, b2, Wih3, Whh3, b3,
      Wih4, Whh4, b4, Wout, bout, (float*)d_out);
}
